// DocRedModel_8899172237499
// MI455X (gfx1250) — matrix operand order repaired and checked by reading, not compiled, not run on hardware
//
#include <hip/hip_runtime.h>
#include <hip/hip_bf16.h>

// ---------------- problem constants ----------------
constexpr int B_   = 4;
constexpr int L_   = 512;
constexpr int H_   = 1024;
constexpr int M_   = 128;
constexpr int E_   = 64;
constexpr int NH_  = 16;
constexpr int R_   = 1024;
constexpr int D_   = 768;
constexpr int C_   = 97;
constexpr int CPAD = 112;              // pad C to multiple of 16
constexpr int NTOT = B_ * R_;          // 4096
constexpr int K2H  = 2 * H_;           // 2048
constexpr long long DB_ = (long long)D_ * 64; // 49152

typedef __bf16 bf16;
typedef __attribute__((ext_vector_type(16))) __bf16 v16bf;
typedef __attribute__((ext_vector_type(8)))  float  v8f;
typedef __attribute__((ext_vector_type(4)))  unsigned uint4v;
typedef __attribute__((ext_vector_type(8)))  unsigned uint8v;

#define WMMA_BF16(acc, a, b) \
  (acc) = __builtin_amdgcn_wmma_f32_16x16x32_bf16(false, (a), false, (b), (short)0, (acc), false, false)

// ---- fragment loaders (CDNA5 16x16x32 bf16 VGPR layouts, ISA 7.12.2) ----
// A (16xK row-major): lanes 0-15 row=lane, K = {k0..k0+7, k0+16..k0+23}
//                     lanes 16-31 row=lane-16, K = {k0+8..k0+15, k0+24..k0+31}
__device__ __forceinline__ v16bf load_a_rowmajor(const bf16* __restrict__ base,
                                                 int ld, int k0, int lane) {
  int row  = lane & 15;
  int koff = (lane & 16) ? 8 : 0;
  const bf16* p = base + (size_t)row * ld + k0 + koff;
  v16bf f;
#pragma unroll
  for (int e = 0; e < 8; ++e) f[e] = p[e];
#pragma unroll
  for (int e = 8; e < 16; ++e) f[e] = p[e + 8];
  return f;
}

// B from pre-transposed BT[N][K] row-major:
// lanes 0-15: col=lane, K={k0..k0+7, k0+16..k0+23} ; lanes 16-31: col=lane-16, K={k0+8..k0+15, k0+24..k0+31}
__device__ __forceinline__ v16bf load_b_transposed(const bf16* __restrict__ baseT,
                                                   int ld, int k0, int lane) {
  int n    = lane & 15;
  int koff = (lane & 16) ? 8 : 0;
  const bf16* p = baseT + (size_t)n * ld + k0 + koff;
  v16bf f;
#pragma unroll
  for (int e = 0; e < 16; ++e) f[e] = p[e < 8 ? e : e + 8];
  return f;
}

// A fragment from an LDS-staged 16x128 chunk (row stride 128 elements)
__device__ __forceinline__ v16bf load_a_lds(const bf16* atile, int ks, int lane) {
  int row  = lane & 15;
  int koff = (lane & 16) ? 8 : 0;
  const bf16* p = atile + row * 128 + ks + koff;
  v16bf f;
#pragma unroll
  for (int e = 0; e < 8; ++e) f[e] = p[e];
#pragma unroll
  for (int e = 8; e < 16; ++e) f[e] = p[e + 8];
  return f;
}

// ---- CDNA5 async staging: 256 threads move a 16x128 bf16 tile (4KB) to LDS ----
__device__ __forceinline__ void async_stage_tile16x128(const bf16* gbase, int ldelem,
                                                       bf16* atile) {
  int t   = threadIdx.x;
  int row = t >> 4;
  int seg = t & 15;
  const bf16* g = gbase + (size_t)row * ldelem + seg * 8;
  // low 32 bits of a generic shared pointer == LDS byte offset
  unsigned ldsoff = (unsigned)(size_t)(atile + row * 128 + seg * 8);
  asm volatile("global_load_async_to_lds_b128 %0, %1, off"
               :: "v"(ldsoff), "v"(g)
               : "memory");
}

// ---- CDNA5 Tensor Data Mover staging: one wave DMAs a 16x128 bf16 tile ----
__device__ __forceinline__ void tdm_stage_tile16x128(const bf16* gbase, int ldelem,
                                                     bf16* atile) {
  unsigned long long ga = (unsigned long long)(size_t)gbase;
  unsigned lds = (unsigned)(size_t)atile;
  uint4v g0;
  g0[0] = 1u;                                                 // count=1, user mode
  g0[1] = lds;                                                // lds_addr
  g0[2] = (unsigned)ga;                                       // global_addr[31:0]
  g0[3] = (unsigned)((ga >> 32) & 0x01FFFFFFu) | (2u << 30);  // addr[56:32] | type=2
  uint8v g1;
  g1[0] = 1u << 16;                                           // data_size = 2 bytes
  g1[1] = ((unsigned)ldelem & 0xFFFFu) << 16;                 // tensor_dim0[15:0]
  g1[2] = (((unsigned)ldelem >> 16) & 0xFFFFu) | (0xFFFFu << 16); // dim0 hi | tensor_dim1 lo
  g1[3] = (128u << 16);                                       // tensor_dim1 hi | tile_dim0=128
  g1[4] = 16u;                                                // tile_dim1=16, tile_dim2=0
  g1[5] = (unsigned)ldelem;                                   // tensor_dim0_stride[31:0]
  g1[6] = 0u;                                                 // stride0 hi | stride1 lo
  g1[7] = 0u;
  asm volatile("tensor_load_to_lds %0, %1"
               :: "s"(g0), "s"(g1)
               : "memory");
  __builtin_amdgcn_s_wait_tensorcnt(0);
}

// ---------------- K0: f32 -> bf16 transpose  dst[n][r] = src[r][n] ----------------
__global__ void k_transpose_bf16(const float* __restrict__ src, bf16* __restrict__ dst,
                                 int rows, int cols, int outRows) {
  size_t idx = (size_t)blockIdx.x * blockDim.x + threadIdx.x;
  size_t tot = (size_t)outRows * rows;
  if (idx >= tot) return;
  int r  = (int)(idx % rows);
  int nn = (int)(idx / rows);
  dst[idx] = (nn < cols) ? (bf16)src[(size_t)r * cols + nn] : (bf16)0.0f;
}

// ---------------- K1: per-entity logsumexp pool + attention mean ----------------
__global__ void k_pool(const float* __restrict__ ent_lhs,   // [B][M][H]
                       const float* __restrict__ attn,      // [B][NH][M][L]
                       const int*   __restrict__ labels,    // [B][M]
                       float* __restrict__ ent_emb,         // [B][E][H]
                       float* __restrict__ ent_attn) {      // [B][E][NH][L]
  int b = blockIdx.x >> 6;
  int e = blockIdx.x & 63;
  __shared__ int mlist[M_];
  __shared__ int cnt_s;
  if (threadIdx.x == 0) cnt_s = 0;
  __syncthreads();
  for (int m = threadIdx.x; m < M_; m += blockDim.x)
    if (labels[b * M_ + m] == e) { int p = atomicAdd(&cnt_s, 1); mlist[p] = m; }
  __syncthreads();
  int cnt = cnt_s;

  const float* eb = ent_lhs + (size_t)b * M_ * H_;
  float* oe = ent_emb + ((size_t)b * E_ + e) * H_;
  for (int h = threadIdx.x; h < H_; h += blockDim.x) {
    float v = 0.0f;
    if (cnt > 0) {
      float mx = -3.4e38f;
      for (int q = 0; q < cnt; ++q) mx = fmaxf(mx, eb[(size_t)mlist[q] * H_ + h]);
      float s = 0.0f;
      for (int q = 0; q < cnt; ++q) s += expf(eb[(size_t)mlist[q] * H_ + h] - mx);
      v = mx + logf(s);
    }
    oe[h] = v;
  }

  const float* ab = attn + (size_t)b * NH_ * M_ * L_;
  float* oa = ent_attn + ((size_t)b * E_ + e) * NH_ * L_;
  float inv = (cnt > 0) ? 1.0f / (float)cnt : 0.0f;
  for (int idx = threadIdx.x; idx < NH_ * L_; idx += blockDim.x) {
    int nh = idx >> 9;
    int l  = idx & (L_ - 1);
    float s = 0.0f;
    for (int q = 0; q < cnt; ++q) s += ab[((size_t)nh * M_ + mlist[q]) * L_ + l];
    oa[idx] = s * inv;
  }
}

// ---------------- K2: pair attention product + gather head/tail embeddings ----------------
__global__ void k_pairs(const int*   __restrict__ hts,       // [B][R][2]
                        const float* __restrict__ ent_emb,   // [B][E][H]
                        const float* __restrict__ ent_attn,  // [B][E][NH][L]
                        bf16* __restrict__ ht_bf,            // [B][R][L]
                        bf16* __restrict__ hcat,             // [N][2H]
                        bf16* __restrict__ tcat) {           // [N][2H]
  int b = blockIdx.x >> 10;
  int r = blockIdx.x & (R_ - 1);
  int n = b * R_ + r;
  int hi = hts[n * 2 + 0];
  int ti = hts[n * 2 + 1];
  const float* ha = ent_attn + ((size_t)b * E_ + hi) * NH_ * L_;
  const float* ta = ent_attn + ((size_t)b * E_ + ti) * NH_ * L_;

  __shared__ float vsh[L_];
  __shared__ float red[256];
  float local = 0.0f;
  for (int l = threadIdx.x; l < L_; l += blockDim.x) {
    float s = 0.0f;
#pragma unroll
    for (int nh = 0; nh < NH_; ++nh) s += ha[nh * L_ + l] * ta[nh * L_ + l];
    s *= (1.0f / NH_);
    vsh[l] = s;
    local += s;
  }
  red[threadIdx.x] = local;
  __syncthreads();
  for (int o = 128; o > 0; o >>= 1) {
    if ((int)threadIdx.x < o) red[threadIdx.x] += red[threadIdx.x + o];
    __syncthreads();
  }
  float scale = 1.0f / (red[0] + 1e-5f);
  for (int l = threadIdx.x; l < L_; l += blockDim.x)
    ht_bf[(size_t)n * L_ + l] = (bf16)(vsh[l] * scale);

  const float* he = ent_emb + ((size_t)b * E_ + hi) * H_;
  const float* te = ent_emb + ((size_t)b * E_ + ti) * H_;
  for (int h = threadIdx.x; h < H_; h += blockDim.x) {
    hcat[(size_t)n * K2H + h] = (bf16)he[h];
    tcat[(size_t)n * K2H + h] = (bf16)te[h];
  }
}

// ---------------- K3: rel = ht_attn @ seq_lhs  (bf16 WMMA, async-LDS staged A) ----------------
// block tile: 16 pairs x 128 hidden; A chunk shared by all 8 waves via async LDS copy
__global__ void k_rel_gemm(const bf16* __restrict__ ht,    // [B][R][L]
                           const bf16* __restrict__ seqT,  // [B][H][L]
                           bf16* __restrict__ hcat, bf16* __restrict__ tcat) {
  __shared__ alignas(16) bf16 atile[16 * 128];
  int wave = threadIdx.x >> 5, lane = threadIdx.x & 31;
  int b    = blockIdx.x >> 9;             // 512 blocks per batch
  int rem  = blockIdx.x & 511;
  int r0   = (rem >> 3) << 4;             // 64 row tiles
  int h0   = ((rem & 7) << 7) + wave * 16;
  const bf16* Abase = ht + ((size_t)b * R_ + r0) * L_;
  const bf16* BT    = seqT + ((size_t)b * H_ + h0) * L_;
  v8f acc = {};
  for (int kc = 0; kc < L_; kc += 128) {
    async_stage_tile16x128(Abase + kc, L_, atile);
    asm volatile("s_wait_asynccnt 0x0" ::: "memory");
    __syncthreads();
#pragma unroll
    for (int ks = 0; ks < 128; ks += 32) {
      v16bf a  = load_a_lds(atile, ks, lane);
      v16bf bf = load_b_transposed(BT, L_, kc + ks, lane);
      WMMA_BF16(acc, a, bf);
    }
    __syncthreads();
  }
  int col = lane & 15;
  int mb  = (lane & 16) ? 8 : 0;
#pragma unroll
  for (int e = 0; e < 8; ++e) {
    size_t n = (size_t)b * R_ + r0 + mb + e;
    size_t c = (size_t)H_ + h0 + col;
    bf16 v = (bf16)acc[e];
    hcat[n * K2H + c] = v;
    tcat[n * K2H + c] = v;
  }
}

// ---------------- K4: hv/tv = tanh(cat @ W + bias)  (bf16 WMMA, TDM staged A) ----------------
// block tile: 16 pairs x 128 dims; A chunk DMA'd to LDS by the Tensor Data Mover
__global__ void k_extract_gemm(const bf16* __restrict__ Acat,  // [N][2H]
                               const bf16* __restrict__ WT,    // [D][2H]
                               const float* __restrict__ bias, // [D]
                               float* __restrict__ outv) {     // [N][D]
  __shared__ alignas(16) bf16 atile[16 * 128];
  int wave = threadIdx.x >> 5, lane = threadIdx.x & 31;
  int n0 = (blockIdx.x / 6) << 4;               // 256 row tiles
  int d0 = (blockIdx.x % 6) * 128 + wave * 16;  // 6 col blocks of 128
  const bf16* Abase = Acat + (size_t)n0 * K2H;
  const bf16* BT    = WT + (size_t)d0 * K2H;
  v8f acc = {};
  for (int kc = 0; kc < K2H; kc += 128) {
    if (threadIdx.x < 32)                       // wave 0 drives the TDM
      tdm_stage_tile16x128(Abase + kc, K2H, atile);
    __syncthreads();
#pragma unroll
    for (int ks = 0; ks < 128; ks += 32) {
      v16bf a  = load_a_lds(atile, ks, lane);
      v16bf bf = load_b_transposed(BT, K2H, kc + ks, lane);
      WMMA_BF16(acc, a, bf);
    }
    __syncthreads();
  }
  int col = lane & 15;
  int mb  = (lane & 16) ? 8 : 0;
  float bsum = bias[d0 + col];
#pragma unroll
  for (int e = 0; e < 8; ++e)
    outv[(size_t)(n0 + mb + e) * D_ + d0 + col] = tanhf(acc[e] + bsum);
}

// ---------------- K5: block-bilinear GEMM  out = (b1 (x) b2) @ Wb + bb ----------------
// A[n][k*4096 + i*64 + j] = hv[n][k*64+i] * tv[n][k*64+j]  built on the fly per fragment
__global__ void k_bilinear_gemm(const float* __restrict__ hv,   // [N][D]
                                const float* __restrict__ tv,   // [N][D]
                                const bf16*  __restrict__ WbT,  // [CPAD][D*64]
                                const float* __restrict__ bbv,  // [C]
                                float* __restrict__ out) {      // [N][C]
  int wave = threadIdx.x >> 5, lane = threadIdx.x & 31;
  int t = blockIdx.x * 8 + wave;                 // 1792 tiles
  int n0 = (t / (CPAD / 16)) << 4;
  int c0 = (t % (CPAD / 16)) << 4;
  int row = lane & 15;
  const float* hvp = hv + (size_t)(n0 + row) * D_;
  const float* tvp = tv + (size_t)(n0 + row) * D_;
  int aoff = (lane & 16) ? 8 : 0;
  const bf16* wrow = WbT + (size_t)(c0 + (lane & 15)) * DB_ + ((lane & 16) ? 8 : 0);
  v8f acc = {};
  for (int s = 0; s < (int)(DB_ / 32); ++s) {
    int kij0 = s << 5;
    int k  = kij0 >> 12;
    int i  = (kij0 >> 6) & 63;
    int j0 = kij0 & 63;                           // 0 or 32
    float hvv = hvp[(k << 6) + i];
    const float* tp = tvp + (k << 6) + j0 + aoff;
    v16bf a;
#pragma unroll
    for (int e = 0; e < 8; ++e)  a[e] = (bf16)(hvv * tp[e]);
#pragma unroll
    for (int e = 8; e < 16; ++e) a[e] = (bf16)(hvv * tp[e + 8]);
    v16bf bw;
#pragma unroll
    for (int e = 0; e < 16; ++e) bw[e] = wrow[kij0 + (e < 8 ? e : e + 8)];
    __builtin_prefetch(wrow + kij0 + 64, 0, 1);   // global_prefetch next chunk
    WMMA_BF16(acc, a, bw);
  }
  int c = c0 + (lane & 15);
  if (c < C_) {
    float bias = bbv[c];
    int mb = (lane & 16) ? 8 : 0;
#pragma unroll
    for (int e = 0; e < 8; ++e)
      out[(size_t)(n0 + mb + e) * C_ + c] = acc[e] + bias;
  }
}

// ---------------- host launch ----------------
extern "C" void kernel_launch(void* const* d_in, const int* in_sizes, int n_in,
                              void* d_out, int out_size, void* d_ws, size_t ws_size,
                              hipStream_t stream) {
  const float* seq_lhs = (const float*)d_in[0];
  const float* ent_lhs = (const float*)d_in[1];
  const float* attn    = (const float*)d_in[2];
  const int*   labels  = (const int*)d_in[3];
  const int*   hts     = (const int*)d_in[4];
  const float* Wh      = (const float*)d_in[5];
  const float* bh      = (const float*)d_in[6];
  const float* Wt      = (const float*)d_in[7];
  const float* bt      = (const float*)d_in[8];
  const float* Wb      = (const float*)d_in[9];
  const float* bb      = (const float*)d_in[10];
  float* out = (float*)d_out;

  char* wsp = (char*)d_ws;
  auto alloc = [&](size_t bytes) -> char* {
    char* p = wsp;
    wsp += (bytes + 255) & ~(size_t)255;
    return p;
  };
  float* ent_emb  = (float*)alloc(sizeof(float) * (size_t)B_ * E_ * H_);
  float* ent_attn = (float*)alloc(sizeof(float) * (size_t)B_ * E_ * NH_ * L_);
  bf16*  ht_bf    = (bf16*) alloc(2ull * B_ * R_ * L_);
  bf16*  seqT     = (bf16*) alloc(2ull * B_ * H_ * L_);
  bf16*  hcat     = (bf16*) alloc(2ull * NTOT * K2H);
  bf16*  tcat     = (bf16*) alloc(2ull * NTOT * K2H);
  bf16*  WhT      = (bf16*) alloc(2ull * D_ * K2H);
  bf16*  WtT      = (bf16*) alloc(2ull * D_ * K2H);
  bf16*  WbT      = (bf16*) alloc(2ull * CPAD * DB_);
  float* hv       = (float*)alloc(4ull * NTOT * D_);
  float* tv       = (float*)alloc(4ull * NTOT * D_);

  // stage transposed bf16 operands
  for (int b = 0; b < B_; ++b)
    k_transpose_bf16<<<(H_ * L_ + 255) / 256, 256, 0, stream>>>(
        seq_lhs + (size_t)b * L_ * H_, seqT + (size_t)b * H_ * L_, L_, H_, H_);
  k_transpose_bf16<<<(D_ * K2H + 255) / 256, 256, 0, stream>>>(Wh, WhT, K2H, D_, D_);
  k_transpose_bf16<<<(D_ * K2H + 255) / 256, 256, 0, stream>>>(Wt, WtT, K2H, D_, D_);
  k_transpose_bf16<<<(int)(((size_t)CPAD * DB_ + 255) / 256), 256, 0, stream>>>(
      Wb, WbT, (int)DB_, C_, CPAD);

  k_pool<<<B_ * E_, 256, 0, stream>>>(ent_lhs, attn, labels, ent_emb, ent_attn);
  k_pairs<<<B_ * R_, 256, 0, stream>>>(hts, ent_emb, ent_attn, ht_bf, hcat, tcat);
  k_rel_gemm<<<B_ * 512, 256, 0, stream>>>(ht_bf, seqT, hcat, tcat);
  k_extract_gemm<<<256 * 6, 256, 0, stream>>>(hcat, WhT, bh, hv);
  k_extract_gemm<<<256 * 6, 256, 0, stream>>>(tcat, WtT, bt, tv);
  k_bilinear_gemm<<<(256 * (CPAD / 16)) / 8, 256, 0, stream>>>(hv, tv, WbT, bb, out);
}
